// DualInputHyperbolicUNet_90056874262973
// MI455X (gfx1250) — matrix operand order repaired and checked by reading, not compiled, not run on hardware
//
#include <hip/hip_runtime.h>
#include <hip/hip_bf16.h>
#include <math.h>

typedef __attribute__((ext_vector_type(16))) _Float16 v16h;
typedef __attribute__((ext_vector_type(8)))  float    v8f;

struct Frag16 {
  union { v16h v; _Float16 h[16]; float4 q[2]; };
};

static constexpr int  TOK   = 2048;   // N = D*H*W
static constexpr int  CCH   = 256;    // channels
static constexpr int  MLPD  = 2048;
static constexpr int  NHEAD = 8;
static constexpr int  HD    = 32;     // head dim
static constexpr float EPS  = 1e-5f;

// ---------------------------------------------------------------------------
// Tensor Data Mover support (guarded: 5-arg builtin on ROCm7.2/clang-22,
// 6-arg on clang-23; cooperative-copy fallback otherwise).
// ---------------------------------------------------------------------------
#if defined(__AMDGCN__) && __has_builtin(__builtin_amdgcn_tensor_load_to_lds) && \
    __has_builtin(__builtin_amdgcn_s_wait_tensorcnt)
#define HAVE_TDM 1
#else
#define HAVE_TDM 0
#endif

#if HAVE_TDM
typedef unsigned int v4u __attribute__((ext_vector_type(4)));
typedef int v4i __attribute__((ext_vector_type(4)));
typedef int v8i __attribute__((ext_vector_type(8)));

// 1-D TDM copy: `bytes` contiguous bytes (multiple of 8) global -> LDS offset.
// D# group0: count=1, lds_addr, 57b global addr, type=2.
// D# group1: data_size=8B, tensor_dim0=1<<20, tile_dim0=bytes/8, tile_dim1=1.
__device__ __forceinline__ void tdm_load_1d(const void* gsrc, unsigned lds_off,
                                            unsigned bytes)
{
  unsigned long long ga = (unsigned long long)(size_t)gsrc;
  unsigned n8 = bytes >> 3;
  v4u g0;
  g0[0] = 1u;                                   // count = 1 (valid user D#)
  g0[1] = lds_off;                              // LDS byte address (first & only
                                                //  __shared__ object => offset 0 base)
  g0[2] = (unsigned)ga;                         // global_addr[31:0]
  g0[3] = (unsigned)(ga >> 32) | (2u << 30);    // global_addr[56:32] | type=2
  v8i g1;
  g1[0] = (int)(3u << 16);                      // data_size = 3 (8 bytes)
  g1[1] = 0;                                    // no atomic barrier; td0[15:0]=0
  g1[2] = (int)(((1u << 20) >> 16) | (1u << 16)); // tensor_dim0 = 1<<20; tensor_dim1=1
  g1[3] = (int)(n8 << 16);                      // tile_dim0 = n8 elements
  g1[4] = 1;                                    // tile_dim1 = 1, tile_dim2 = 0
  g1[5] = (int)n8;                              // tensor_dim0_stride
  g1[6] = 0;
  g1[7] = 0;
  v4i z4 = {0, 0, 0, 0};
#if __clang_major__ >= 23
  v8i z8 = {0, 0, 0, 0, 0, 0, 0, 0};
  __builtin_amdgcn_tensor_load_to_lds(g0, g1, z4, z4, z8, 0);
#else
  __builtin_amdgcn_tensor_load_to_lds(g0, g1, z4, z4, 0);
#endif
}
#endif

// ---------------------------------------------------------------------------
// Repack fp32 weight W (K x Nc, row-major) into fp16 B-fragment order:
//   Wp[((kt*NT + nt)*32 + lane)*16 + e] = W[kt*32 + (lane>>4)*16 + e][nt*16 + (lane&15)]
// One k-step x 64-col panel is a contiguous 4KB chunk (TDM/LDS friendly).
// ---------------------------------------------------------------------------
__global__ void repack_w(const float* __restrict__ W, _Float16* __restrict__ Wp,
                         int K, int Nc)
{
  int idx = blockIdx.x * blockDim.x + threadIdx.x;
  if (idx >= K * Nc) return;
  int k  = idx / Nc, nn = idx - k * Nc;
  int kt = k >> 5, kr = k & 31;
  int nt = nn >> 4, nr = nn & 15;
  int lane = ((kr >> 4) << 4) | nr;
  int e    = kr & 15;
  Wp[(((size_t)kt * (Nc >> 4) + nt) * 32 + lane) * 16 + e] = (_Float16)W[idx];
}

// ---------------------------------------------------------------------------
// WMMA GEMM: out(M x Nc) = A(M x K f16 row-major) * Wp(packed) + bias.
// 128-thread (4-wave) workgroup computes a 64x64 tile; the 4KB B panel per
// 32-wide k-step is staged in LDS (double-buffered, TDM async when available)
// and shared by all 4 waves (4x L2 traffic reduction on weights).
// ---------------------------------------------------------------------------
template<int FUSE_GELU, typename OutT>
__global__ void __launch_bounds__(128)
gemm16(const _Float16* __restrict__ A, const _Float16* __restrict__ Wp,
       const float* __restrict__ bias, OutT* __restrict__ out,
       int M, int K, int Nc)
{
  __shared__ __align__(16) _Float16 bsh[2][2048];   // 2 x 4KB B panels

  const int tid  = threadIdx.x;
  const int lane = tid & 31;
  const int wave = tid >> 5;
  const int nlo = lane & 15, hi = lane >> 4;
  const int mt  = blockIdx.x;           // 64-row tile
  const int nt0 = blockIdx.y << 2;      // 4 x 16 columns
  const int NT  = Nc >> 4;
  const int KT  = K >> 5;

  const _Float16* arow = A + (size_t)(mt * 64 + wave * 16 + nlo) * K + hi * 16;

  auto stage = [&](int kt, int dbuf) {
    const _Float16* gsrc = Wp + ((size_t)kt * NT + nt0) * 512;
#if HAVE_TDM
    if (tid < 32) tdm_load_1d(gsrc, (unsigned)(dbuf * 4096), 4096u);
#else
    ((float4*)&bsh[dbuf][0])[tid]       = *(const float4*)(gsrc + (size_t)tid * 8);
    ((float4*)&bsh[dbuf][0])[tid + 128] = *(const float4*)(gsrc + (size_t)(tid + 128) * 8);
#endif
  };
  auto wait_stage = [&]() {
#if HAVE_TDM
    if (tid < 32) __builtin_amdgcn_s_wait_tensorcnt(0);
#endif
  };

  v8f acc[4] = {};

  stage(0, 0);
  wait_stage();
  __syncthreads();

  for (int kt = 0; kt < KT; ++kt) {
    int buf = kt & 1;
    if (kt + 1 < KT) stage(kt + 1, buf ^ 1);
    if (kt + 2 < KT) __builtin_prefetch(arow + (kt + 2) * 32, 0, 0);

    Frag16 a;
    a.q[0] = *(const float4*)(arow + kt * 32);
    a.q[1] = *(const float4*)(arow + kt * 32 + 8);
    const _Float16* bp = &bsh[buf][(size_t)lane * 16];
#pragma unroll
    for (int t = 0; t < 4; ++t) {
      Frag16 b;
      b.q[0] = *(const float4*)(bp + t * 512);
      b.q[1] = *(const float4*)(bp + t * 512 + 8);
      acc[t] = __builtin_amdgcn_wmma_f32_16x16x32_f16(
          false, a.v, false, b.v, (short)0, acc[t], false, false);
    }
    wait_stage();
    __syncthreads();
  }

#pragma unroll
  for (int t = 0; t < 4; ++t) {
    int col = (nt0 + t) * 16 + nlo;
    float bv = bias ? bias[col] : 0.0f;
#pragma unroll
    for (int r = 0; r < 8; ++r) {
      int row = mt * 64 + wave * 16 + r + 8 * hi;   // C-layout: VGPR r -> M=r+8*hi
      float v = acc[t][r] + bv;
      if (FUSE_GELU) v = 0.5f * v * (1.0f + erff(v * 0.70710678118654752f));
      out[(size_t)row * Nc + col] = (OutT)v;
    }
  }
}

// ---------------------------------------------------------------------------
// LayerNorm over C for input stored (C, N); emits fp16 (N, C).
// ---------------------------------------------------------------------------
__global__ void __launch_bounds__(256)
ln_in(const float* __restrict__ x, const float* __restrict__ g,
      const float* __restrict__ b, _Float16* __restrict__ xn, int N, int C)
{
  __shared__ float red[256];
  int n = blockIdx.x, c = threadIdx.x;
  float v = x[(size_t)c * N + n];
  red[c] = v; __syncthreads();
  for (int s = 128; s > 0; s >>= 1) { if (c < s) red[c] += red[c + s]; __syncthreads(); }
  float mu = red[0] * (1.0f / 256.0f);
  __syncthreads();
  float d = v - mu;
  red[c] = d * d; __syncthreads();
  for (int s = 128; s > 0; s >>= 1) { if (c < s) red[c] += red[c + s]; __syncthreads(); }
  float var = red[0] * (1.0f / 256.0f);
  xn[(size_t)n * C + c] = (_Float16)(d * rsqrtf(var + EPS) * g[c] + b[c]);
}

// proj (N,C) + (wA+wB)*bias + residual (read transposed from (C,N) input),
// then LayerNorm -> fp16 (N,C). Also stores fp32 residual sum.
__global__ void __launch_bounds__(256)
res_ln(const float* __restrict__ proj, const float* __restrict__ bo,
       const float* __restrict__ wA, const float* __restrict__ wB,
       const float* __restrict__ xin, const float* __restrict__ g,
       const float* __restrict__ beta, float* __restrict__ resv,
       _Float16* __restrict__ xn2, int N, int C)
{
  __shared__ float red[256];
  int n = blockIdx.x, c = threadIdx.x;
  float ws = *wA + *wB;
  float v = proj[(size_t)n * C + c] + ws * bo[c] + xin[(size_t)c * N + n];
  resv[(size_t)n * C + c] = v;
  red[c] = v; __syncthreads();
  for (int s = 128; s > 0; s >>= 1) { if (c < s) red[c] += red[c + s]; __syncthreads(); }
  float mu = red[0] * (1.0f / 256.0f);
  __syncthreads();
  float d = v - mu;
  red[c] = d * d; __syncthreads();
  for (int s = 128; s > 0; s >>= 1) { if (c < s) red[c] += red[c + s]; __syncthreads(); }
  float var = red[0] * (1.0f / 256.0f);
  xn2[(size_t)n * C + c] = (_Float16)(d * rsqrtf(var + EPS) * g[c] + beta[c]);
}

// vt[c][key] = v[key][c] (fp16 transpose so V^T A-fragments load contiguously)
__global__ void transpose_nc(const _Float16* __restrict__ v,
                             _Float16* __restrict__ vt, int N, int C)
{
  int idx = blockIdx.x * blockDim.x + threadIdx.x;
  if (idx >= N * C) return;
  int key = idx / C, c = idx - key * C;
  vt[(size_t)c * N + key] = v[idx];
}

// o = wA*a + wB*b  (fp16), scalars read from device
__global__ void combine_f16(const _Float16* __restrict__ a, const _Float16* __restrict__ b,
                            const float* __restrict__ wA, const float* __restrict__ wB,
                            _Float16* __restrict__ o, int n)
{
  int i = blockIdx.x * blockDim.x + threadIdx.x;
  if (i >= n) return;
  o[i] = (_Float16)((*wA) * (float)a[i] + (*wB) * (float)b[i]);
}

// out(C,N) = m2x + xres + m2y + yres   (all (N,C)) with transposed store
__global__ void final_out(const float* __restrict__ m2x, const float* __restrict__ xr,
                          const float* __restrict__ m2y, const float* __restrict__ yr,
                          float* __restrict__ out, int N, int C)
{
  int i = blockIdx.x * blockDim.x + threadIdx.x;
  if (i >= N * C) return;
  int n = i / C, c = i - n * C;
  out[(size_t)c * N + n] = m2x[i] + xr[i] + m2y[i] + yr[i];
}

// ---------------------------------------------------------------------------
// Flash-style attention, one wave per (head, 16-query block).
// S^T = K*Q^T so each lane owns one query column -> lane-local online-softmax
// stats (one shfl_xor(16) merge). O^T = V^T * P^T as two 16x16 fragments.
// P^T (C-layout) -> B-fragment uses 16 packed-f16 ds_bpermute per 32 keys.
// ---------------------------------------------------------------------------
__global__ void __launch_bounds__(32)
attn_wmma(const _Float16* __restrict__ Q, const _Float16* __restrict__ K,
          const _Float16* __restrict__ VT, _Float16* __restrict__ O, int N)
{
  const int C = CCH;
  const int lane = threadIdx.x;
  const int nlo = lane & 15, hi = lane >> 4;
  const int head = blockIdx.x >> 7;           // N/16 = 128 query blocks
  const int q0 = (blockIdx.x & 127) << 4;
  const float scale = 0.17677669529663687f;   // 1/sqrt(32)

  Frag16 qf;  // Q^T B-fragment: lane = query q0+nlo, dims hi*8+e / 16+hi*8+(e-8) (table order)
  {
    const _Float16* qp = Q + (size_t)(q0 + nlo) * C + head * HD + hi * 8;
    qf.q[0] = *(const float4*)qp;
    qf.q[1] = *(const float4*)(qp + 16);
  }

  v8f o0 = {}, o1 = {};
  float rmax = -3.0e38f, rsum = 0.0f;

  for (int j = 0; j < N; j += 32) {
    Frag16 ka0, ka1;   // K A-fragments (keys j..j+15 / j+16..j+31)
    const _Float16* kp = K + (size_t)(j + nlo) * C + head * HD + hi * 8;
    ka0.q[0] = *(const float4*)(kp);
    ka0.q[1] = *(const float4*)(kp + 16);
    kp += 16 * C;
    ka1.q[0] = *(const float4*)(kp);
    ka1.q[1] = *(const float4*)(kp + 16);

    v8f s0 = {}, s1 = {};
    s0 = __builtin_amdgcn_wmma_f32_16x16x32_f16(false, ka0.v, false, qf.v, (short)0, s0, false, false);
    s1 = __builtin_amdgcn_wmma_f32_16x16x32_f16(false, ka1.v, false, qf.v, (short)0, s1, false, false);

    float p0f[8], p1f[8];
    float lm = -3.0e38f;
#pragma unroll
    for (int r = 0; r < 8; ++r) {
      p0f[r] = s0[r] * scale;
      p1f[r] = s1[r] * scale;
      lm = fmaxf(lm, fmaxf(p0f[r], p1f[r]));
    }
    lm = fmaxf(lm, __shfl_xor(lm, 16, 32));      // merge partner half-rows
    float nm = fmaxf(rmax, lm);
    float corr = __expf(rmax - nm);
    rmax = nm;

    float ls = 0.0f;
    unsigned pk[8];                               // (tile0, tile1) f16 pair per row
#pragma unroll
    for (int r = 0; r < 8; ++r) {
      float e0 = __expf(p0f[r] - nm);
      float e1 = __expf(p1f[r] - nm);
      ls += e0 + e1;
      union { _Float16 h[2]; unsigned u; } cv;
      cv.h[0] = (_Float16)e0;
      cv.h[1] = (_Float16)e1;
      pk[r] = cv.u;
    }
    ls += __shfl_xor(ls, 16, 32);
    rsum = rsum * corr + ls;
#pragma unroll
    for (int r = 0; r < 8; ++r) { o0[r] *= corr; o1[r] *= corr; }

    // P^T (C-layout) -> B-fragment: dest lane (col nlo, half hi) needs key row
    // hi*16+e; row e lives in lane nlo (e<8, VGPR e) or nlo+16 (e>=8, VGPR e-8).
    Frag16 pb;
#pragma unroll
    for (int e = 0; e < 16; ++e) {
      int src = nlo + ((e & 8) ? 16 : 0);
      union { unsigned u; _Float16 h[2]; } cv;
      cv.u = __shfl(pk[e & 7], src, 32);
      pb.h[e] = cv.h[hi];                        // hi selects 16-key tile
    }

    Frag16 va0, va1;   // V^T A-fragments: lane dim = nlo (+16), contiguous keys
    const _Float16* vp0 = VT + (size_t)(head * HD + nlo) * N + j + hi * 16;
    va0.q[0] = *(const float4*)(vp0);
    va0.q[1] = *(const float4*)(vp0 + 8);
    const _Float16* vp1 = vp0 + 16 * N;
    va1.q[0] = *(const float4*)(vp1);
    va1.q[1] = *(const float4*)(vp1 + 8);

    o0 = __builtin_amdgcn_wmma_f32_16x16x32_f16(false, va0.v, false, pb.v, (short)0, o0, false, false);
    o1 = __builtin_amdgcn_wmma_f32_16x16x32_f16(false, va1.v, false, pb.v, (short)0, o1, false, false);
  }

  float inv = 1.0f / rsum;
  union H8 { _Float16 h[8]; float4 f4; } w0, w1;
#pragma unroll
  for (int r = 0; r < 8; ++r) {
    w0.h[r] = (_Float16)(o0[r] * inv);   // dims r + 8*hi
    w1.h[r] = (_Float16)(o1[r] * inv);   // dims 16 + r + 8*hi
  }
  _Float16* op = O + (size_t)(q0 + nlo) * C + head * HD + hi * 8;
  *(float4*)(op)      = w0.f4;
  *(float4*)(op + 16) = w1.f4;
}

// ---------------------------------------------------------------------------
// Host orchestration
// ---------------------------------------------------------------------------
extern "C" void kernel_launch(void* const* d_in, const int* in_sizes, int n_in,
                              void* d_out, int out_size, void* d_ws, size_t ws_size,
                              hipStream_t stream)
{
  (void)in_sizes; (void)n_in; (void)out_size; (void)ws_size;
  const int N = TOK, C = CCH, M = MLPD;

  const float* x = (const float*)d_in[0];
  const float* y = (const float*)d_in[1];
  const float* ln_x1_g = (const float*)d_in[2];
  const float* ln_x1_b = (const float*)d_in[3];
  const float* ln_y1_g = (const float*)d_in[4];
  const float* ln_y1_b = (const float*)d_in[5];
  const float* ln_x2_g = (const float*)d_in[6];
  const float* ln_x2_b = (const float*)d_in[7];
  const float* ln_y2_g = (const float*)d_in[8];
  const float* ln_y2_b = (const float*)d_in[9];
  const float* Wsrc[8] = { (const float*)d_in[10], (const float*)d_in[12],
                           (const float*)d_in[14], (const float*)d_in[16],
                           (const float*)d_in[18], (const float*)d_in[20],
                           (const float*)d_in[22], (const float*)d_in[24] };
  const float* bq_x = (const float*)d_in[11];
  const float* bk_x = (const float*)d_in[13];
  const float* bv_x = (const float*)d_in[15];
  const float* bq_y = (const float*)d_in[17];
  const float* bk_y = (const float*)d_in[19];
  const float* bv_y = (const float*)d_in[21];
  const float* bo_x = (const float*)d_in[23];
  const float* bo_y = (const float*)d_in[25];
  const float* W1x = (const float*)d_in[26];
  const float* b1x = (const float*)d_in[27];
  const float* W2x = (const float*)d_in[28];
  const float* b2x = (const float*)d_in[29];
  const float* W1y = (const float*)d_in[30];
  const float* b1y = (const float*)d_in[31];
  const float* W2y = (const float*)d_in[32];
  const float* b2y = (const float*)d_in[33];
  const float* w11 = (const float*)d_in[34];
  const float* w12 = (const float*)d_in[35];
  const float* w21 = (const float*)d_in[36];
  const float* w22 = (const float*)d_in[37];

  size_t off = 0;
  auto alloc = [&](size_t bytes) -> void* {
    size_t a = (off + 255) & ~(size_t)255;
    off = a + bytes;
    return (void*)((char*)d_ws + a);
  };

  _Float16* wp[8];
  for (int i = 0; i < 8; ++i) wp[i] = (_Float16*)alloc((size_t)C * C * 2);
  _Float16* wp1x = (_Float16*)alloc((size_t)C * M * 2);
  _Float16* wp2x = (_Float16*)alloc((size_t)M * C * 2);
  _Float16* wp1y = (_Float16*)alloc((size_t)C * M * 2);
  _Float16* wp2y = (_Float16*)alloc((size_t)M * C * 2);
  _Float16* xn  = (_Float16*)alloc((size_t)N * C * 2);
  _Float16* yn  = (_Float16*)alloc((size_t)N * C * 2);
  _Float16* qx  = (_Float16*)alloc((size_t)N * C * 2);
  _Float16* kx  = (_Float16*)alloc((size_t)N * C * 2);
  _Float16* vx  = (_Float16*)alloc((size_t)N * C * 2);
  _Float16* qy  = (_Float16*)alloc((size_t)N * C * 2);
  _Float16* ky  = (_Float16*)alloc((size_t)N * C * 2);
  _Float16* vy  = (_Float16*)alloc((size_t)N * C * 2);
  _Float16* vtx = (_Float16*)alloc((size_t)N * C * 2);
  _Float16* vty = (_Float16*)alloc((size_t)N * C * 2);
  _Float16* aox  = (_Float16*)alloc((size_t)N * C * 2);
  _Float16* aoy  = (_Float16*)alloc((size_t)N * C * 2);
  _Float16* aocx = (_Float16*)alloc((size_t)N * C * 2);
  _Float16* aocy = (_Float16*)alloc((size_t)N * C * 2);
  _Float16* cmbx = (_Float16*)alloc((size_t)N * C * 2);
  _Float16* cmby = (_Float16*)alloc((size_t)N * C * 2);
  float* projx = (float*)alloc((size_t)N * C * 4);
  float* projy = (float*)alloc((size_t)N * C * 4);
  float* xattn = (float*)alloc((size_t)N * C * 4);
  float* yattn = (float*)alloc((size_t)N * C * 4);
  _Float16* xn2 = (_Float16*)alloc((size_t)N * C * 2);
  _Float16* yn2 = (_Float16*)alloc((size_t)N * C * 2);
  _Float16* hx  = (_Float16*)alloc((size_t)N * M * 2);
  _Float16* hy  = (_Float16*)alloc((size_t)N * M * 2);
  float* m2x = (float*)alloc((size_t)N * C * 4);
  float* m2y = (float*)alloc((size_t)N * C * 4);

  // 1) repack weights fp32 -> fp16 B-fragment layout
  for (int i = 0; i < 8; ++i)
    repack_w<<<(C * C + 255) / 256, 256, 0, stream>>>(Wsrc[i], wp[i], C, C);
  repack_w<<<(C * M + 255) / 256, 256, 0, stream>>>(W1x, wp1x, C, M);
  repack_w<<<(M * C + 255) / 256, 256, 0, stream>>>(W2x, wp2x, M, C);
  repack_w<<<(C * M + 255) / 256, 256, 0, stream>>>(W1y, wp1y, C, M);
  repack_w<<<(M * C + 255) / 256, 256, 0, stream>>>(W2y, wp2y, M, C);

  // 2) LN1
  ln_in<<<N, 256, 0, stream>>>(x, ln_x1_g, ln_x1_b, xn, N, C);
  ln_in<<<N, 256, 0, stream>>>(y, ln_y1_g, ln_y1_b, yn, N, C);

  // 3) QKV projections  (64x64 tile per 128-thread block)
  dim3 gCC(N / 64, C / 64);
  gemm16<0, _Float16><<<gCC, 128, 0, stream>>>(xn, wp[0], bq_x, qx, N, C, C);
  gemm16<0, _Float16><<<gCC, 128, 0, stream>>>(xn, wp[1], bk_x, kx, N, C, C);
  gemm16<0, _Float16><<<gCC, 128, 0, stream>>>(xn, wp[2], bv_x, vx, N, C, C);
  gemm16<0, _Float16><<<gCC, 128, 0, stream>>>(yn, wp[3], bq_y, qy, N, C, C);
  gemm16<0, _Float16><<<gCC, 128, 0, stream>>>(yn, wp[4], bk_y, ky, N, C, C);
  gemm16<0, _Float16><<<gCC, 128, 0, stream>>>(yn, wp[5], bv_y, vy, N, C, C);

  // 4) transpose V for V^T A-fragment loads
  transpose_nc<<<(N * C + 255) / 256, 256, 0, stream>>>(vx, vtx, N, C);
  transpose_nc<<<(N * C + 255) / 256, 256, 0, stream>>>(vy, vty, N, C);

  // 5) four attentions (self-x, self-y, cross x->y, cross y->x)
  int ablk = NHEAD * (N / 16);
  attn_wmma<<<ablk, 32, 0, stream>>>(qx, kx, vtx, aox,  N);
  attn_wmma<<<ablk, 32, 0, stream>>>(qy, ky, vty, aoy,  N);
  attn_wmma<<<ablk, 32, 0, stream>>>(qx, ky, vty, aocx, N);
  attn_wmma<<<ablk, 32, 0, stream>>>(qy, kx, vtx, aocy, N);

  // 6) fold the two projections sharing Wo into one GEMM each
  combine_f16<<<(N * C + 255) / 256, 256, 0, stream>>>(aox, aocx, w11, w12, cmbx, N * C);
  combine_f16<<<(N * C + 255) / 256, 256, 0, stream>>>(aoy, aocy, w21, w22, cmby, N * C);
  gemm16<0, float><<<gCC, 128, 0, stream>>>(cmbx, wp[6], nullptr, projx, N, C, C);
  gemm16<0, float><<<gCC, 128, 0, stream>>>(cmby, wp[7], nullptr, projy, N, C, C);

  // 7) bias + residual + LN2 (fused)
  res_ln<<<N, 256, 0, stream>>>(projx, bo_x, w11, w12, x, ln_x2_g, ln_x2_b, xattn, xn2, N, C);
  res_ln<<<N, 256, 0, stream>>>(projy, bo_y, w21, w22, y, ln_y2_g, ln_y2_b, yattn, yn2, N, C);

  // 8) MLP (GELU fused into GEMM1 epilogue)
  dim3 gCM(N / 64, M / 64);
  gemm16<1, _Float16><<<gCM, 128, 0, stream>>>(xn2, wp1x, b1x, hx, N, C, M);
  gemm16<1, _Float16><<<gCM, 128, 0, stream>>>(yn2, wp1y, b1y, hy, N, C, M);
  gemm16<0, float><<<gCC, 128, 0, stream>>>(hx, wp2x, b2x, m2x, N, M, C);
  gemm16<0, float><<<gCC, 128, 0, stream>>>(hy, wp2y, b2y, m2y, N, M, C);

  // 9) fused residual sum + transpose back to (C, N)
  final_out<<<(N * C + 255) / 256, 256, 0, stream>>>(m2x, xattn, m2y, yattn,
                                                     (float*)d_out, N, C);
}
